// EnhancedMoEModel_206158430469
// MI455X (gfx1250) — matrix operand order repaired and checked by reading, not compiled, not run on hardware
//
#include <hip/hip_runtime.h>
#include <stdint.h>

// ---------------------------------------------------------------------------
// Soft-cluster MoE inference: out[b] = sum_e sigmoid(MLP_e(x[b])) * probs[b,e]
// B=524288, D=32, H=64, H2=32, O=1, E=8.
// bf16 WMMA (V_WMMA_F32_16X16X32_BF16) for stage1/stage2 matmuls.
// Load-all / wmma-all / epilogue-all scheduling to hide WMMA->VALU hazards.
// Wave-private LDS bounce with s_wait_dscnt fences.
// ---------------------------------------------------------------------------

typedef __attribute__((ext_vector_type(16))) __bf16 v16bf;
typedef __attribute__((ext_vector_type(2)))  __bf16 v2bf;
typedef __attribute__((ext_vector_type(8)))  float  v8f;

#define E_  8
#define D_  32
#define H_  64
#define H2_ 32

// LDS layout (bytes), all 16B aligned
#define OFF_W1F   0          // 32 frags * 1024B  (8 experts * 4 n-blocks)
#define OFF_W2F   32768      // 32 frags * 1024B  (8 experts * 2 k-blocks * 2 n-blocks)
#define OFF_B1    65536      // 512 f32
#define OFF_B2    67584      // 256 f32
#define OFF_W3    68608      // 256 f32
#define OFF_B3    69632      // 8 f32 (padded to 64B)
#define OFF_H1    69696      // 8 waves * 2048B : bf16 h1 tile [16][64]
#define OFF_H2    86080      // 8 waves * 2048B : f32  h2 tile [16][32]
#define LDS_TOTAL 102464

// f32 pair -> packed bf16 dword. Use plain __bf16 casts (fptrunc) so the
// backend can select a hardware packed convert if gfx1250 has one.
#if __has_builtin(__builtin_amdgcn_cvt_pk_bf16_f32)
static __device__ inline unsigned pk_bf16(float a, float b) {
  union { v2bf v; unsigned u; } r;
  r.v = __builtin_amdgcn_cvt_pk_bf16_f32(a, b);
  return r.u;
}
#else
static __device__ inline unsigned pk_bf16(float a, float b) {
  union { v2bf v; unsigned u; } r;
  r.v[0] = (__bf16)a;
  r.v[1] = (__bf16)b;
  return r.u;
}
#endif

// Wave-local LDS RAW fence: LDS ops of one wave are tracked by DScnt and
// issued in order; waiting DScnt==0 guarantees prior ds_stores landed before
// subsequent ds_loads. "memory" clobber pins compiler ordering.
static __device__ inline void lds_fence_wave() {
  asm volatile("s_wait_dscnt 0x0" ::: "memory");
}

union FragAB { uint4 q[2]; unsigned u[8]; v16bf v; };

__global__ __launch_bounds__(256)
void moe_soft_cluster_wmma(const float* __restrict__ x,
                           const float* __restrict__ probs,
                           const float* __restrict__ W1,
                           const float* __restrict__ b1,
                           const float* __restrict__ W2,
                           const float* __restrict__ b2,
                           const float* __restrict__ W3,
                           const float* __restrict__ b3,
                           float* __restrict__ out,
                           int tilesPerWave)
{
  extern __shared__ char smem[];
  unsigned* w1f = (unsigned*)(smem + OFF_W1F);
  unsigned* w2f = (unsigned*)(smem + OFF_W2F);
  float* sb1 = (float*)(smem + OFF_B1);
  float* sb2 = (float*)(smem + OFF_B2);
  float* sw3 = (float*)(smem + OFF_W3);
  float* sb3 = (float*)(smem + OFF_B3);

  const int tid  = threadIdx.x;
  const int lane = tid & 31;
  const int wave = tid >> 5;

  // ---------------- pre-swizzle weights into WMMA B-fragment layout --------
  // B-frag (32x16, bf16): lane l -> N = l&15, kq = (l<16 ? 0 : 8),
  // dword j holds K = kq+2(j&3)+16(j>>2) (lo half) and that K+1 (hi half).
  for (int idx = tid; idx < 8192; idx += 256) {        // W1: fid = e*4 + nb
    int fid = idx >> 8, r = idx & 255;
    int l = r >> 3, j = r & 7;
    int e = fid >> 2, nb = fid & 3;
    int n = l & 15, kq = (l < 16) ? 0 : 8;
    int k0  = kq + 2 * (j & 3) + (j >> 2) * 16;
    int col = nb * 16 + n;
    w1f[idx] = pk_bf16(W1[(e * D_ + k0) * H_ + col],
                       W1[(e * D_ + k0 + 1) * H_ + col]);
  }
  for (int idx = tid; idx < 8192; idx += 256) {        // W2: fid = e*4 + kb*2 + nb
    int fid = idx >> 8, r = idx & 255;
    int l = r >> 3, j = r & 7;
    int e = fid >> 2, kb = (fid >> 1) & 1, nb = fid & 1;
    int n = l & 15, kq = (l < 16) ? 0 : 8;
    int k   = kb * 32 + kq + 2 * (j & 3) + (j >> 2) * 16;
    int col = nb * 16 + n;
    w2f[idx] = pk_bf16(W2[(e * H_ + k) * H2_ + col],
                       W2[(e * H_ + k + 1) * H2_ + col]);
  }
  for (int idx = tid; idx < 512; idx += 256) sb1[idx] = b1[idx];
  if (tid < 256) sb2[tid] = b2[tid];
  if (tid < 256) sw3[tid] = W3[tid];
  if (tid < 8)   sb3[tid] = b3[tid];
  __syncthreads();   // w1f/w2f shared across waves: real workgroup barrier

  uint16_t* h1w = (uint16_t*)(smem + OFF_H1 + wave * 2048);
  float*    h2w = (float*)   (smem + OFF_H2 + wave * 2048);

  const int hi  = lane >> 4;          // 0 or 1 (half-wave)
  const int nlo = lane & 15;
  const int b0  = hi ? 8 : 0;         // A-frag K sub-offset per half-wave

  const long waveGlobal = (long)blockIdx.x * 8 + wave;

  for (int t = 0; t < tilesPerWave; ++t) {
    const long tile = waveGlobal * tilesPerWave + t;
    const long row0 = tile * 16;

    // --- A-fragment of x (16x32 bf16): lane l -> M = l&15,
    //     dwords 0-3: K = b0+0..7, dwords 4-7: K = 16+b0..16+b0+7
    FragAB Ax;
    {
      const float* xr = x + (row0 + nlo) * D_;
      float4 f0 = *(const float4*)(xr + b0);
      float4 f1 = *(const float4*)(xr + b0 + 4);
      float4 f2 = *(const float4*)(xr + 16 + b0);
      float4 f3 = *(const float4*)(xr + 16 + b0 + 4);
      Ax.u[0] = pk_bf16(f0.x, f0.y); Ax.u[1] = pk_bf16(f0.z, f0.w);
      Ax.u[2] = pk_bf16(f1.x, f1.y); Ax.u[3] = pk_bf16(f1.z, f1.w);
      Ax.u[4] = pk_bf16(f2.x, f2.y); Ax.u[5] = pk_bf16(f2.z, f2.w);
      Ax.u[6] = pk_bf16(f3.x, f3.y); Ax.u[7] = pk_bf16(f3.z, f3.w);
    }

    float oacc = 0.0f;

    #pragma unroll 1
    for (int e = 0; e < E_; ++e) {
      // ---------------- stage 1: h1 = relu(x @ W1[e] + b1[e]) -------------
      // load all 4 B-fragments, fire all 4 WMMAs back-to-back (no hazard
      // NOPs between independent WMMAs), then run all epilogues.
      FragAB Bw[4];
      #pragma unroll
      for (int nb = 0; nb < 4; ++nb) {
        const uint4* p = (const uint4*)(w1f + ((e * 4 + nb) * 256 + lane * 8));
        Bw[nb].q[0] = p[0]; Bw[nb].q[1] = p[1];
      }
      v8f c1[4];
      #pragma unroll
      for (int nb = 0; nb < 4; ++nb) {
        v8f z = {};
        c1[nb] = __builtin_amdgcn_wmma_f32_16x16x32_bf16(false, Ax.v, false,
                                                         Bw[nb].v, (short)0, z,
                                                         false, false);
      }
      #pragma unroll
      for (int nb = 0; nb < 4; ++nb) {
        const int col = nb * 16 + nlo;
        const float bv = sb1[e * H_ + col];
        #pragma unroll
        for (int r = 0; r < 8; r += 2) {
          float h0 = fmaxf(c1[nb][r]     + bv, 0.f);
          float h1 = fmaxf(c1[nb][r + 1] + bv, 0.f);
          unsigned d = pk_bf16(h0, h1);
          h1w[(r     + hi * 8) * H_ + col] = (uint16_t)(d & 0xFFFFu);
          h1w[(r + 1 + hi * 8) * H_ + col] = (uint16_t)(d >> 16);
        }
      }
      lds_fence_wave();   // wave-private RAW: h1 stores -> h1 A-frag loads

      // ---------------- stage 2: h2 = relu(h1 @ W2[e] + b2[e]) ------------
      FragAB A0, A1;   // A-frags of h1: K=0..31 and K=32..63
      A0.q[0] = *(const uint4*)(h1w + (nlo * H_ + b0));
      A0.q[1] = *(const uint4*)(h1w + (nlo * H_ + 16 + b0));
      A1.q[0] = *(const uint4*)(h1w + (nlo * H_ + 32 + b0));
      A1.q[1] = *(const uint4*)(h1w + (nlo * H_ + 48 + b0));
      FragAB B2[4];    // index kb*2 + nb
      #pragma unroll
      for (int q = 0; q < 4; ++q) {
        const uint4* p = (const uint4*)(w2f + ((e * 4 + q) * 256 + lane * 8));
        B2[q].q[0] = p[0]; B2[q].q[1] = p[1];
      }
      v8f z0 = {}, z1 = {};
      v8f d0 = __builtin_amdgcn_wmma_f32_16x16x32_bf16(false, A0.v, false,
                                                       B2[0].v, (short)0, z0,
                                                       false, false);
      v8f d1 = __builtin_amdgcn_wmma_f32_16x16x32_bf16(false, A0.v, false,
                                                       B2[1].v, (short)0, z1,
                                                       false, false);
      d0 = __builtin_amdgcn_wmma_f32_16x16x32_bf16(false, A1.v, false,
                                                   B2[2].v, (short)0, d0,
                                                   false, false);
      d1 = __builtin_amdgcn_wmma_f32_16x16x32_bf16(false, A1.v, false,
                                                   B2[3].v, (short)0, d1,
                                                   false, false);
      {
        const float bv0 = sb2[e * H2_ + nlo];
        const float bv1 = sb2[e * H2_ + 16 + nlo];
        #pragma unroll
        for (int r = 0; r < 8; ++r) {
          h2w[(r + hi * 8) * H2_ + nlo]      = fmaxf(d0[r] + bv0, 0.f);
          h2w[(r + hi * 8) * H2_ + 16 + nlo] = fmaxf(d1[r] + bv1, 0.f);
        }
      }
      lds_fence_wave();   // wave-private RAW: h2 stores -> h2 loads

      // ---------------- stage 3: sigmoid(h2 @ W3[e] + b3[e]) * probs ------
      if (lane < 16) {
        const int m = lane;
        float s = sb3[e];
        const float* w3e = sw3 + e * H2_;
        #pragma unroll
        for (int k = 0; k < H2_; k += 4) {
          float4 hv = *(const float4*)(h2w + m * H2_ + k);
          s += hv.x * w3e[k] + hv.y * w3e[k + 1] + hv.z * w3e[k + 2] + hv.w * w3e[k + 3];
        }
        float sig = 1.0f / (1.0f + __expf(-s));
        oacc += sig * probs[(row0 + m) * E_ + e];
      }
      // WAR (next expert h1 stores vs this expert's reads) is safe: LDS ops
      // of a wave issue and complete in order (DScnt, ISA 15.5).
    }

    if (lane < 16) out[row0 + lane] = oacc;
  }
}

extern "C" void kernel_launch(void* const* d_in, const int* in_sizes, int n_in,
                              void* d_out, int out_size, void* d_ws, size_t ws_size,
                              hipStream_t stream) {
  (void)n_in; (void)d_ws; (void)ws_size; (void)out_size;
  const float* x  = (const float*)d_in[0];
  const float* pr = (const float*)d_in[1];
  const float* W1 = (const float*)d_in[2];
  const float* b1 = (const float*)d_in[3];
  const float* W2 = (const float*)d_in[4];
  const float* b2 = (const float*)d_in[5];
  const float* W3 = (const float*)d_in[6];
  const float* b3 = (const float*)d_in[7];
  float* out = (float*)d_out;

  const int B = in_sizes[0] / D_;           // 524288
  const int blocks = 1024;                  // 8 waves each
  const int tilesPerWave = B / (16 * 8 * blocks);   // 4 for B=524288

  moe_soft_cluster_wmma<<<blocks, 256, LDS_TOTAL, stream>>>(
      x, pr, W1, b1, W2, b2, W3, b3, out, tilesPerWave);
}
